// DEQTransformerLM_70274254897383
// MI455X (gfx1250) — matrix operand order repaired and checked by reading, not compiled, not run on hardware
//
#include <hip/hip_runtime.h>
#include <hip/hip_bf16.h>
#include <stdint.h>

// Problem constants (from reference)
#define BATCH 4
#define DMODEL 1024
#define QLN 512
#define MLN 512
#define KLN 1024
#define NHEAD 16
#define DHEAD 64
#define DINNER 4096
#define TDM3 3072   // 3*d_model

typedef __attribute__((ext_vector_type(16))) __bf16 v16bf;
typedef __attribute__((ext_vector_type(8)))  float  v8f;
typedef __attribute__((ext_vector_type(16))) unsigned short v16us;
typedef __attribute__((ext_vector_type(8)))  unsigned short v8us;

__device__ __forceinline__ unsigned short f2bf(float f) {
  unsigned int u = __builtin_bit_cast(unsigned int, f);
  unsigned int r = (u + 0x7FFFu + ((u >> 16) & 1u)) >> 16;  // round-nearest-even
  return (unsigned short)r;
}
__device__ __forceinline__ float bf2f(unsigned short h) {
  unsigned int u = ((unsigned int)h) << 16;
  return __builtin_bit_cast(float, u);
}

// ---- WMMA fragment loads (wave32, 16x16x32 bf16) ----
// A (16xK, row-major, K-contiguous): lane holds row lane&15,
//   16 contiguous K at (lane>>4)*16.
__device__ __forceinline__ v16bf load_frag_a(const unsigned short* A, int lda,
                                             int row0, int k0, int lane) {
  const unsigned short* p = A + (size_t)(row0 + (lane & 15)) * lda + k0 + ((lane >> 4) << 4);
  v8us lo = *(const v8us*)p;
  v8us hi = *(const v8us*)(p + 8);
  v16us t;
#pragma unroll
  for (int e = 0; e < 8; ++e) { t[e] = lo[e]; t[e + 8] = hi[e]; }
  return __builtin_bit_cast(v16bf, t);
}
// B (Kx16) given as BT row-major (col-major B, K-contiguous): lane holds
//   column lane&15, 16 contiguous K at (lane>>4)*16.
__device__ __forceinline__ v16bf load_frag_b(const unsigned short* BT, int ldb,
                                             int col0, int k0, int lane) {
  const unsigned short* p = BT + (size_t)(col0 + (lane & 15)) * ldb + k0 + ((lane >> 4) << 4);
  v16us t = *(const v16us*)p;
  return __builtin_bit_cast(v16bf, t);
}
__device__ __forceinline__ v8f wmma_bf16(v16bf a, v16bf b, v8f c) {
  return __builtin_amdgcn_wmma_f32_16x16x32_bf16(false, a, false, b, (short)0, c, false, false);
}

// ---- 32x32-per-wave GEMM core ----
__device__ __forceinline__ void gemm32x32(const unsigned short* A, int lda,
                                          const unsigned short* BT, int ldb, int K,
                                          int m0, int n0, int lane,
                                          v8f& c00, v8f& c01, v8f& c10, v8f& c11) {
  for (int k0 = 0; k0 < K; k0 += 32) {
    v16bf a0 = load_frag_a(A, lda, m0, k0, lane);
    v16bf a1 = load_frag_a(A, lda, m0 + 16, k0, lane);
    v16bf b0 = load_frag_b(BT, ldb, n0, k0, lane);
    v16bf b1 = load_frag_b(BT, ldb, n0 + 16, k0, lane);
    c00 = wmma_bf16(a0, b0, c00);
    c01 = wmma_bf16(a0, b1, c01);
    c10 = wmma_bf16(a1, b0, c10);
    c11 = wmma_bf16(a1, b1, c11);
  }
}

// ---- conversion / layout kernels ----
__global__ void f32_to_bf16_k(const float* __restrict__ in, unsigned short* __restrict__ out, int n) {
  int i = blockIdx.x * 256 + threadIdx.x;
  if (i < n) out[i] = f2bf(in[i]);
}
// catT[b][k][d] = (k<512? z0[b][d][k] : z1ss[b][d][k-512]), bf16
__global__ void build_catT_k(const float* __restrict__ z0, const float* __restrict__ z1,
                             unsigned short* __restrict__ catT) {
  int idx = blockIdx.x * 256 + threadIdx.x;
  if (idx >= BATCH * KLN * DMODEL) return;
  int d = idx & (DMODEL - 1);
  int k = (idx >> 10) & (KLN - 1);
  int b = idx >> 20;
  float v = (k < MLN) ? z0[((size_t)b * DMODEL + d) * MLN + k]
                      : z1[((size_t)b * DMODEL + d) * QLN + (k - MLN)];
  catT[idx] = f2bf(v);
}
// posT[k][d] = pos_emb[0][d][k]
__global__ void build_posT_k(const float* __restrict__ pos, unsigned short* __restrict__ posT) {
  int idx = blockIdx.x * 256 + threadIdx.x;
  if (idx >= KLN * DMODEL) return;
  int d = idx & (DMODEL - 1);
  int k = idx >> 10;
  posT[idx] = f2bf(pos[(size_t)d * KLN + k]);
}
// rw_qT/rr_qT[b][n][i][d] from w_headsT (q block, last 512 cols) + biases
__global__ void prep_q_k(const unsigned short* __restrict__ whT,
                         const float* __restrict__ rwb, const float* __restrict__ rrb,
                         unsigned short* __restrict__ rwqT, unsigned short* __restrict__ rrqT) {
  int idx = blockIdx.x * 256 + threadIdx.x;
  if (idx >= BATCH * NHEAD * QLN * DHEAD) return;
  int d = idx & 63;
  int i = (idx >> 6) & 511;
  int n = (idx >> 15) & 15;
  int b = idx >> 19;
  float q = bf2f(whT[((size_t)b * KLN + (MLN + i)) * TDM3 + n * 64 + d]);
  rwqT[idx] = f2bf(q + rwb[n * 64 + d]);
  rrqT[idx] = f2bf(q + rrb[n * 64 + d]);
}
// v_nat[b][n][d][j] = w_headsT[b][j][2048 + n*64 + d]   (plain bf16 transpose-copy)
__global__ void prep_v_k(const unsigned short* __restrict__ whT, unsigned short* __restrict__ vnat) {
  int idx = blockIdx.x * 256 + threadIdx.x;
  if (idx >= BATCH * NHEAD * DHEAD * KLN) return;
  int j = idx & (KLN - 1);
  int d = (idx >> 10) & 63;
  int n = (idx >> 16) & 15;
  int b = idx >> 20;
  vnat[idx] = whT[((size_t)b * KLN + j) * TDM3 + 2048 + n * 64 + d];
}

// ---- GEMM kernels (128x64 macro tile, 8 waves of 32x32) ----
// w_headsT[b][k][o] = bf16( qkv_w @ cat + uss )
__global__ __launch_bounds__(256) void gemm_qkv_k(
    const unsigned short* __restrict__ A, const unsigned short* __restrict__ catT,
    const float* __restrict__ uss, unsigned short* __restrict__ whT) {
  int lane = threadIdx.x & 31, w = threadIdx.x >> 5;
  int m0 = blockIdx.x * 128 + (w & 3) * 32;
  int n0 = blockIdx.y * 64 + (w >> 2) * 32;
  int b = blockIdx.z;
  const unsigned short* BT = catT + (size_t)b * KLN * DMODEL;
  v8f z = {};
  v8f c00 = z, c01 = z, c10 = z, c11 = z;
  gemm32x32(A, DMODEL, BT, DMODEL, DMODEL, m0, n0, lane, c00, c01, c10, c11);
  int rr = (lane >> 4) << 3, cc = lane & 15;
#pragma unroll
  for (int mi = 0; mi < 2; ++mi)
#pragma unroll
    for (int ni = 0; ni < 2; ++ni) {
      v8f acc = mi == 0 ? (ni == 0 ? c00 : c01) : (ni == 0 ? c10 : c11);
      int k = n0 + ni * 16 + cc;
#pragma unroll
      for (int j = 0; j < 8; ++j) {
        int o = m0 + mi * 16 + rr + j;
        float v = acc[j] + uss[((size_t)b * TDM3 + o) * KLN + k];
        whT[((size_t)b * KLN + k) * TDM3 + o] = f2bf(v);
      }
    }
}
// rT[k][o] = bf16( r_w @ pos_emb ) transposed
__global__ __launch_bounds__(256) void gemm_r_k(
    const unsigned short* __restrict__ A, const unsigned short* __restrict__ posT,
    unsigned short* __restrict__ rT) {
  int lane = threadIdx.x & 31, w = threadIdx.x >> 5;
  int m0 = blockIdx.x * 128 + (w & 3) * 32;
  int n0 = blockIdx.y * 64 + (w >> 2) * 32;
  v8f z = {};
  v8f c00 = z, c01 = z, c10 = z, c11 = z;
  gemm32x32(A, DMODEL, posT, DMODEL, DMODEL, m0, n0, lane, c00, c01, c10, c11);
  int rr = (lane >> 4) << 3, cc = lane & 15;
#pragma unroll
  for (int mi = 0; mi < 2; ++mi)
#pragma unroll
    for (int ni = 0; ni < 2; ++ni) {
      v8f acc = mi == 0 ? (ni == 0 ? c00 : c01) : (ni == 0 ? c10 : c11);
      int k = n0 + ni * 16 + cc;
#pragma unroll
      for (int j = 0; j < 8; ++j) {
        int o = m0 + mi * 16 + rr + j;
        rT[(size_t)k * DMODEL + o] = f2bf(acc[j]);
      }
    }
}
// pre1[b][o][i] = o_w @ attn_vec + o_b + z1ss
__global__ __launch_bounds__(256) void gemm_o_k(
    const unsigned short* __restrict__ A, const unsigned short* __restrict__ avT,
    const float* __restrict__ ob, const float* __restrict__ z1ss, float* __restrict__ pre1) {
  int lane = threadIdx.x & 31, w = threadIdx.x >> 5;
  int m0 = blockIdx.x * 128 + (w & 3) * 32;
  int n0 = blockIdx.y * 64 + (w >> 2) * 32;
  int b = blockIdx.z;
  const unsigned short* BT = avT + (size_t)b * QLN * DMODEL;
  v8f z = {};
  v8f c00 = z, c01 = z, c10 = z, c11 = z;
  gemm32x32(A, DMODEL, BT, DMODEL, DMODEL, m0, n0, lane, c00, c01, c10, c11);
  int rr = (lane >> 4) << 3, cc = lane & 15;
#pragma unroll
  for (int mi = 0; mi < 2; ++mi)
#pragma unroll
    for (int ni = 0; ni < 2; ++ni) {
      v8f acc = mi == 0 ? (ni == 0 ? c00 : c01) : (ni == 0 ? c10 : c11);
      int i = n0 + ni * 16 + cc;
#pragma unroll
      for (int j = 0; j < 8; ++j) {
        int o = m0 + mi * 16 + rr + j;
        size_t oi = ((size_t)b * DMODEL + o) * QLN + i;
        pre1[oi] = acc[j] + ob[o] + z1ss[oi];
      }
    }
}
// hT[b][i][o] = bf16( relu( ff1_w @ x + ff1_b ) )  stored transposed
__global__ __launch_bounds__(256) void gemm_ff1_k(
    const unsigned short* __restrict__ A, const unsigned short* __restrict__ xT,
    const float* __restrict__ f1b, unsigned short* __restrict__ hT) {
  int lane = threadIdx.x & 31, w = threadIdx.x >> 5;
  int m0 = blockIdx.x * 128 + (w & 3) * 32;
  int n0 = blockIdx.y * 64 + (w >> 2) * 32;
  int b = blockIdx.z;
  const unsigned short* BT = xT + (size_t)b * QLN * DMODEL;
  v8f z = {};
  v8f c00 = z, c01 = z, c10 = z, c11 = z;
  gemm32x32(A, DMODEL, BT, DMODEL, DMODEL, m0, n0, lane, c00, c01, c10, c11);
  int rr = (lane >> 4) << 3, cc = lane & 15;
#pragma unroll
  for (int mi = 0; mi < 2; ++mi)
#pragma unroll
    for (int ni = 0; ni < 2; ++ni) {
      v8f acc = mi == 0 ? (ni == 0 ? c00 : c01) : (ni == 0 ? c10 : c11);
      int i = n0 + ni * 16 + cc;
#pragma unroll
      for (int j = 0; j < 8; ++j) {
        int o = m0 + mi * 16 + rr + j;
        float v = fmaxf(acc[j] + f1b[o], 0.f);
        hT[((size_t)b * QLN + i) * DINNER + o] = f2bf(v);
      }
    }
}
// pre2[b][o][i] = ff2_w @ h + ff2_b + x
__global__ __launch_bounds__(256) void gemm_ff2_k(
    const unsigned short* __restrict__ A, const unsigned short* __restrict__ hT,
    const float* __restrict__ f2b, const float* __restrict__ xf, float* __restrict__ pre2) {
  int lane = threadIdx.x & 31, w = threadIdx.x >> 5;
  int m0 = blockIdx.x * 128 + (w & 3) * 32;
  int n0 = blockIdx.y * 64 + (w >> 2) * 32;
  int b = blockIdx.z;
  const unsigned short* BT = hT + (size_t)b * QLN * DINNER;
  v8f z = {};
  v8f c00 = z, c01 = z, c10 = z, c11 = z;
  gemm32x32(A, DINNER, BT, DINNER, DINNER, m0, n0, lane, c00, c01, c10, c11);
  int rr = (lane >> 4) << 3, cc = lane & 15;
#pragma unroll
  for (int mi = 0; mi < 2; ++mi)
#pragma unroll
    for (int ni = 0; ni < 2; ++ni) {
      v8f acc = mi == 0 ? (ni == 0 ? c00 : c01) : (ni == 0 ? c10 : c11);
      int i = n0 + ni * 16 + cc;
#pragma unroll
      for (int j = 0; j < 8; ++j) {
        int o = m0 + mi * 16 + rr + j;
        size_t oi = ((size_t)b * DMODEL + o) * QLN + i;
        pre2[oi] = acc[j] + f2b[o] + xf[oi];
      }
    }
}

// ---- fused local-window attention: AC + shifted BD -> softmax -> P@V ----
// One WG = (b,n, 64-query tile); key window = [i0+256, i0+576), 320 keys.
// BD rel_shift: BD[i,j] = BDpre[i, j-i+511]; needed cols always r in [768,1024).
#define AT_SAC (64 * 320)
#define AT_SBD (64 * 256)
#define AT_SP  (64 * 320)
#define ATTN_SMEM_BYTES (AT_SAC * 4 + AT_SBD * 4 + AT_SP * 2)

__global__ __launch_bounds__(256) void attn_k(
    const unsigned short* __restrict__ rwqT, const unsigned short* __restrict__ rrqT,
    const unsigned short* __restrict__ whT, const unsigned short* __restrict__ rT,
    const unsigned short* __restrict__ vnat, unsigned short* __restrict__ avT) {
  extern __shared__ char smem_raw[];
  float* sAC = (float*)smem_raw;                               // [64][320]
  float* sBD = (float*)(smem_raw + AT_SAC * 4);                // [64][256]
  unsigned short* sP = (unsigned short*)(smem_raw + AT_SAC * 4 + AT_SBD * 4);  // [64][320]

  int bn = blockIdx.x, b = bn >> 4, n = bn & 15;
  int i0 = blockIdx.y * 64;
  int lane = threadIdx.x & 31, w = threadIdx.x >> 5;
  int rb = w & 3;   // 16-query row block
  int wj = w >> 2;  // column-half owner

  const unsigned short* qrw = rwqT + ((size_t)bn * QLN + i0) * DHEAD;
  const unsigned short* qrr = rrqT + ((size_t)bn * QLN + i0) * DHEAD;
  const unsigned short* kT = whT + (size_t)b * KLN * TDM3 + (DMODEL + n * DHEAD);
  const unsigned short* rTh = rT + n * DHEAD;
  const unsigned short* vh = vnat + (size_t)bn * DHEAD * KLN;

  // Phase 1: AC tiles (64x320) -> LDS
  {
    v16bf a0 = load_frag_a(qrw, DHEAD, rb * 16, 0, lane);
    v16bf a1 = load_frag_a(qrw, DHEAD, rb * 16, 32, lane);
    int rrw = rb * 16 + ((lane >> 4) << 3), cc = lane & 15;
    for (int t = 0; t < 10; ++t) {
      int jt = wj * 10 + t;
      int jcol = i0 + 256 + jt * 16;
      v8f acc = {};
      v16bf b0 = load_frag_b(kT, TDM3, jcol, 0, lane);
      v16bf b1 = load_frag_b(kT, TDM3, jcol, 32, lane);
      acc = wmma_bf16(a0, b0, acc);
      acc = wmma_bf16(a1, b1, acc);
#pragma unroll
      for (int j = 0; j < 8; ++j) sAC[(rrw + j) * 320 + jt * 16 + cc] = acc[j];
    }
  }
  // Phase 2: BDpre tiles (64x256, r in [768,1024)) -> LDS
  {
    v16bf a0 = load_frag_a(qrr, DHEAD, rb * 16, 0, lane);
    v16bf a1 = load_frag_a(qrr, DHEAD, rb * 16, 32, lane);
    int rrw = rb * 16 + ((lane >> 4) << 3), cc = lane & 15;
    for (int t = 0; t < 8; ++t) {
      int rt = wj * 8 + t;
      v8f acc = {};
      v16bf b0 = load_frag_b(rTh, DMODEL, 768 + rt * 16, 0, lane);
      v16bf b1 = load_frag_b(rTh, DMODEL, 768 + rt * 16, 32, lane);
      acc = wmma_bf16(a0, b0, acc);
      acc = wmma_bf16(a1, b1, acc);
#pragma unroll
      for (int j = 0; j < 8; ++j) sBD[(rrw + j) * 256 + rt * 16 + cc] = acc[j];
    }
  }
  __syncthreads();
  // Phase 3: masked softmax per query row (valid jj in [i+1, i+256])
  if (threadIdx.x < 64) {
    int i = threadIdx.x;
    float mx = -3.4e38f;
    for (int jj = i + 1; jj <= i + 256; ++jj) {
      float raw = sAC[i * 320 + jj] + sBD[i * 256 + (jj - i - 1)];
      mx = fmaxf(mx, raw);
    }
    float sum = 0.f;
    for (int jj = i + 1; jj <= i + 256; ++jj) {
      float raw = sAC[i * 320 + jj] + sBD[i * 256 + (jj - i - 1)];
      float e = __expf(0.125f * (raw - mx));  // scale = 1/sqrt(64)
      sAC[i * 320 + jj] = e;
      sum += e;
    }
    float inv = 1.f / sum;
    for (int jj = 0; jj < 320; ++jj) {
      float p = (jj > i && jj <= i + 256) ? sAC[i * 320 + jj] * inv : 0.f;
      sP[i * 320 + jj] = f2bf(p);
    }
  }
  __syncthreads();
  // Phase 4: avT[i][d] = P (64x320) @ V^T window; store bf16 [b][i][n*64+d]
  {
    int rrw = rb * 16 + ((lane >> 4) << 3), cc = lane & 15;
    for (int t = 0; t < 2; ++t) {
      int dt = wj * 2 + t;
      v8f acc = {};
      for (int ks = 0; ks < 10; ++ks) {
        v16bf a = load_frag_a(sP, 320, rb * 16, ks * 32, lane);
        v16bf bb = load_frag_b(vh, KLN, dt * 16, i0 + 256 + ks * 32, lane);
        acc = wmma_bf16(a, bb, acc);
      }
#pragma unroll
      for (int j = 0; j < 8; ++j) {
        int qi = i0 + rrw + j;
        avT[((size_t)b * QLN + qi) * DMODEL + n * DHEAD + dt * 16 + cc] = f2bf(acc[j]);
      }
    }
  }
}

// ---- LayerNorm over channel dim (one thread per (b, i) column) ----
__global__ void ln1_k(const float* __restrict__ pre, float* __restrict__ xf,
                      unsigned short* __restrict__ xT) {
  int idx = blockIdx.x * blockDim.x + threadIdx.x;
  if (idx >= BATCH * QLN) return;
  int b = idx >> 9, i = idx & (QLN - 1);
  const float* p = pre + (size_t)b * DMODEL * QLN + i;
  float s = 0.f, s2 = 0.f;
  for (int o = 0; o < DMODEL; ++o) { float v = p[(size_t)o * QLN]; s += v; s2 += v * v; }
  float mean = s * (1.f / DMODEL);
  float var = s2 * (1.f / DMODEL) - mean * mean;
  float inv = rsqrtf(var + 1e-5f);
  float* xo = xf + (size_t)b * DMODEL * QLN + i;
  unsigned short* xt = xT + ((size_t)b * QLN + i) * DMODEL;
  for (int o = 0; o < DMODEL; ++o) {
    float v = (p[(size_t)o * QLN] - mean) * inv;
    xo[(size_t)o * QLN] = v;
    xt[o] = f2bf(v);
  }
}
__global__ void ln2_k(const float* __restrict__ pre, float* __restrict__ out) {
  int idx = blockIdx.x * blockDim.x + threadIdx.x;
  if (idx >= BATCH * QLN) return;
  int b = idx >> 9, i = idx & (QLN - 1);
  const float* p = pre + (size_t)b * DMODEL * QLN + i;
  float s = 0.f, s2 = 0.f;
  for (int o = 0; o < DMODEL; ++o) { float v = p[(size_t)o * QLN]; s += v; s2 += v * v; }
  float mean = s * (1.f / DMODEL);
  float var = s2 * (1.f / DMODEL) - mean * mean;
  float inv = rsqrtf(var + 1e-5f);
  float* q = out + (size_t)b * DMODEL * QLN + i;
  for (int o = 0; o < DMODEL; ++o) q[(size_t)o * QLN] = (p[(size_t)o * QLN] - mean) * inv;
}

extern "C" void kernel_launch(void* const* d_in, const int* in_sizes, int n_in,
                              void* d_out, int out_size, void* d_ws, size_t ws_size,
                              hipStream_t stream) {
  (void)in_sizes; (void)n_in; (void)out_size; (void)ws_size;
  const float* z1ss = (const float*)d_in[0];
  const float* uss  = (const float*)d_in[1];
  const float* z0   = (const float*)d_in[2];
  const float* pos  = (const float*)d_in[3];
  const float* qkvw = (const float*)d_in[4];
  const float* rw   = (const float*)d_in[5];
  const float* rwb  = (const float*)d_in[6];
  const float* rrb  = (const float*)d_in[7];
  const float* ow   = (const float*)d_in[8];
  const float* ob   = (const float*)d_in[9];
  const float* f1w  = (const float*)d_in[10];
  const float* f1b  = (const float*)d_in[11];
  const float* f2w  = (const float*)d_in[12];
  const float* f2b  = (const float*)d_in[13];
  float* out = (float*)d_out;

  char* ws = (char*)d_ws;
  size_t off = 0;
  auto alloc = [&](size_t bytes) -> char* {
    char* p = ws + off;
    off += (bytes + 255) & ~(size_t)255;
    return p;
  };
  unsigned short* qkvw_bf = (unsigned short*)alloc((size_t)TDM3 * DMODEL * 2);
  unsigned short* rw_bf   = (unsigned short*)alloc((size_t)DMODEL * DMODEL * 2);
  unsigned short* ow_bf   = (unsigned short*)alloc((size_t)DMODEL * DMODEL * 2);
  unsigned short* f1w_bf  = (unsigned short*)alloc((size_t)DINNER * DMODEL * 2);
  unsigned short* f2w_bf  = (unsigned short*)alloc((size_t)DMODEL * DINNER * 2);
  unsigned short* catT    = (unsigned short*)alloc((size_t)BATCH * KLN * DMODEL * 2);
  unsigned short* posT    = (unsigned short*)alloc((size_t)KLN * DMODEL * 2);
  unsigned short* whT     = (unsigned short*)alloc((size_t)BATCH * KLN * TDM3 * 2);
  unsigned short* rT      = (unsigned short*)alloc((size_t)KLN * DMODEL * 2);
  unsigned short* rwqT    = (unsigned short*)alloc((size_t)BATCH * NHEAD * QLN * DHEAD * 2);
  unsigned short* rrqT    = (unsigned short*)alloc((size_t)BATCH * NHEAD * QLN * DHEAD * 2);
  unsigned short* vnat    = (unsigned short*)alloc((size_t)BATCH * NHEAD * DHEAD * KLN * 2);
  unsigned short* avT     = (unsigned short*)alloc((size_t)BATCH * QLN * DMODEL * 2);
  float* pre1             = (float*)alloc((size_t)BATCH * DMODEL * QLN * 4);
  float* xf               = (float*)alloc((size_t)BATCH * DMODEL * QLN * 4);
  unsigned short* xT      = (unsigned short*)alloc((size_t)BATCH * QLN * DMODEL * 2);
  unsigned short* hT      = (unsigned short*)alloc((size_t)BATCH * QLN * DINNER * 2);
  float* pre2             = (float*)alloc((size_t)BATCH * DMODEL * QLN * 4);

  auto cdiv = [](int a, int b) { return (a + b - 1) / b; };

  // Stage inputs to bf16 layouts
  f32_to_bf16_k<<<cdiv(TDM3 * DMODEL, 256), 256, 0, stream>>>(qkvw, qkvw_bf, TDM3 * DMODEL);
  f32_to_bf16_k<<<cdiv(DMODEL * DMODEL, 256), 256, 0, stream>>>(rw, rw_bf, DMODEL * DMODEL);
  f32_to_bf16_k<<<cdiv(DMODEL * DMODEL, 256), 256, 0, stream>>>(ow, ow_bf, DMODEL * DMODEL);
  f32_to_bf16_k<<<cdiv(DINNER * DMODEL, 256), 256, 0, stream>>>(f1w, f1w_bf, DINNER * DMODEL);
  f32_to_bf16_k<<<cdiv(DMODEL * DINNER, 256), 256, 0, stream>>>(f2w, f2w_bf, DMODEL * DINNER);
  build_catT_k<<<cdiv(BATCH * KLN * DMODEL, 256), 256, 0, stream>>>(z0, z1ss, catT);
  build_posT_k<<<cdiv(KLN * DMODEL, 256), 256, 0, stream>>>(pos, posT);

  // QKV projection (+uss) and rel-pos projection
  gemm_qkv_k<<<dim3(TDM3 / 128, KLN / 64, BATCH), 256, 0, stream>>>(qkvw_bf, catT, uss, whT);
  gemm_r_k<<<dim3(DMODEL / 128, KLN / 64, 1), 256, 0, stream>>>(rw_bf, posT, rT);

  // Head-layout prep
  prep_q_k<<<cdiv(BATCH * NHEAD * QLN * DHEAD, 256), 256, 0, stream>>>(whT, rwb, rrb, rwqT, rrqT);
  prep_v_k<<<cdiv(BATCH * NHEAD * DHEAD * KLN, 256), 256, 0, stream>>>(whT, vnat);

  // Fused local attention
  attn_k<<<dim3(BATCH * NHEAD, QLN / 64), 256, ATTN_SMEM_BYTES, stream>>>(rwqT, rrqT, whT, rT,
                                                                          vnat, avT);

  // Output projection + residual, LN1
  gemm_o_k<<<dim3(DMODEL / 128, QLN / 64, BATCH), 256, 0, stream>>>(ow_bf, avT, ob, z1ss, pre1);
  ln1_k<<<cdiv(BATCH * QLN, 256), 256, 0, stream>>>(pre1, xf, xT);

  // FFN + residual, LN2 -> out
  gemm_ff1_k<<<dim3(DINNER / 128, QLN / 64, BATCH), 256, 0, stream>>>(f1w_bf, xT, f1b, hT);
  gemm_ff2_k<<<dim3(DMODEL / 128, QLN / 64, BATCH), 256, 0, stream>>>(f2w_bf, hT, f2b, xf, pre2);
  ln2_k<<<cdiv(BATCH * QLN, 256), 256, 0, stream>>>(pre2, out);
}
